// Conv1D_71760313582388
// MI455X (gfx1250) — compile-verified, matrix operand order checked by reading, not run on hardware
//
#include <hip/hip_runtime.h>

// ---------------------------------------------------------------------------
// Dequant-GEMM for MI455X (gfx1250, wave32, WMMA):
//   y[m,n] = (sum_k x[m,k] * w_q[k,n]) * scale[n] + bias[n]
// scale folded into the epilogue; inner product runs in bf16 WMMA, f32 accum.
// M=8192, N=16384, K=4096.
// ---------------------------------------------------------------------------

typedef __attribute__((ext_vector_type(16))) __bf16 v16bf;
typedef __attribute__((ext_vector_type(8)))  float  v8f;

#define BM 128
#define BN 128
#define KT 32      // one V_WMMA_F32_16X16X32_BF16 K-step per tile

union FragAB { v16bf f; uint4 q[2]; };          // 32B fragment (8 VGPRs)
union BF4    { __bf16 h[4];  uint2 u; };        // 4 bf16 = 8B
union BF16x  { __bf16 h[16]; uint4 q[2]; };     // 16 bf16 = 32B

__global__ __launch_bounds__(256)
void conv1d_wmma_bf16(const float* __restrict__ x,
                      const int*   __restrict__ wq,
                      const float* __restrict__ scale,
                      const float* __restrict__ bias,
                      float*       __restrict__ out,
                      int M, int N, int K)
{
    // Double-buffered tiles: A row-major [m][k], B transposed [n][k] so every
    // WMMA fragment is two contiguous ds_load_b128 per lane.
    __shared__ __align__(16) __bf16 As[2][BM * KT];   // 8 KB each
    __shared__ __align__(16) __bf16 Bs[2][BN * KT];   // 8 KB each

    const int tid  = threadIdx.x;
    const int bx   = blockIdx.x;     // N-tile index
    const int by   = blockIdx.y;     // M-tile index

    const int lane = tid & 31;       // wave32
    const int wave = tid >> 5;       // 8 waves per block
    const int wr   = wave >> 1;      // 0..3 : wave's 32-row band
    const int wc   = wave & 1;       // 0..1 : wave's 64-col band
    const int lm   = lane & 15;
    const int half = lane >> 4;      // lanes 16-31 take the upper K-halves

    // --- cooperative global->LDS staging assignments -----------------------
    const int ar = tid >> 3;         // A: row 0..31 (x4 passes covers 128)
    const int ac = (tid & 7) * 4;    // A: k-col 0,4,...,28 (float4)
    const int bn = tid & 127;        // B: column n within tile
    const int bk = (tid >> 7) * 16;  // B: k 0..15 or 16..31

    const float* xBase = x  + (size_t)(by * BM) * K;
    const int*   wBase = wq + (size_t)(bx * BN) + bn;

    float4 aReg[4];
    int    bReg[16];
    const int NKT = K / KT;

    auto loadGlobal = [&](int kt) {
#pragma unroll
        for (int p = 0; p < 4; ++p)
            aReg[p] = *(const float4*)(xBase + (size_t)(ar + 32 * p) * K
                                             + (size_t)kt * KT + ac);
        const int* wp = wBase + (size_t)(kt * KT + bk) * N;
#pragma unroll
        for (int j = 0; j < 16; ++j)
            bReg[j] = wp[(size_t)j * N];        // coalesced across lanes
    };

    auto stageLDS = [&](int buf) {
#pragma unroll
        for (int p = 0; p < 4; ++p) {
            BF4 t;
            t.h[0] = (__bf16)aReg[p].x;
            t.h[1] = (__bf16)aReg[p].y;
            t.h[2] = (__bf16)aReg[p].z;
            t.h[3] = (__bf16)aReg[p].w;
            *(uint2*)&As[buf][(ar + 32 * p) * KT + ac] = t.u;   // ds_store_b64
        }
        BF16x t;                                // int -> bf16 exact (|w|<=127)
#pragma unroll
        for (int j = 0; j < 16; ++j)
            t.h[j] = (__bf16)(float)bReg[j];
        *(uint4*)&Bs[buf][bn * KT + bk]     = t.q[0];           // ds_store_b128
        *(uint4*)&Bs[buf][bn * KT + bk + 8] = t.q[1];
    };

    v8f zero = {0.f, 0.f, 0.f, 0.f, 0.f, 0.f, 0.f, 0.f};
    v8f acc[2][4];
#pragma unroll
    for (int mt = 0; mt < 2; ++mt)
#pragma unroll
        for (int nt = 0; nt < 4; ++nt)
            acc[mt][nt] = zero;

    // --- software-pipelined main loop (double buffer, one barrier/iter) ----
    loadGlobal(0);
    int buf = 0;
    for (int kt = 0; kt < NKT; ++kt) {
        stageLDS(buf);
        __syncthreads();
        if (kt + 1 < NKT) loadGlobal(kt + 1);   // prefetch next tile to VGPRs

        // Fragment gathers: ISA 16-bit A 16x32 layout -> lane (m=lm, half)
        // needs K[half*8 .. half*8+7] and K[16+half*8 ..] : 2x ds_load_b128.
        FragAB a[2], b[4];
#pragma unroll
        for (int mt = 0; mt < 2; ++mt) {
            const __bf16* p = &As[buf][(wr * 32 + mt * 16 + lm) * KT + half * 8];
            a[mt].q[0] = *(const uint4*)p;
            a[mt].q[1] = *(const uint4*)(p + 16);
        }
#pragma unroll
        for (int nt = 0; nt < 4; ++nt) {
            const __bf16* p = &Bs[buf][(wc * 64 + nt * 16 + lm) * KT + half * 8];
            b[nt].q[0] = *(const uint4*)p;
            b[nt].q[1] = *(const uint4*)(p + 16);
        }

#pragma unroll
        for (int mt = 0; mt < 2; ++mt)
#pragma unroll
            for (int nt = 0; nt < 4; ++nt)
                acc[mt][nt] = __builtin_amdgcn_wmma_f32_16x16x32_bf16(
                    /*neg_a=*/false, a[mt].f,
                    /*neg_b=*/false, b[nt].f,
                    /*c_mod=*/(short)0, acc[mt][nt],
                    /*reuse_a=*/false, /*reuse_b=*/false);

        buf ^= 1;
    }

    // --- epilogue: acc * scale[n] + bias[n] --------------------------------
    // C/D layout: VGPR i, lanes 0-15 -> (M=i, N=lane); lanes 16-31 -> (M=i+8).
#pragma unroll
    for (int nt = 0; nt < 4; ++nt) {
        const int ng = bx * BN + wc * 64 + nt * 16 + lm;
        const float s  = scale[ng];
        const float bb = bias[ng];
#pragma unroll
        for (int mt = 0; mt < 2; ++mt) {
#pragma unroll
            for (int i = 0; i < 8; ++i) {
                const int mg = by * BM + wr * 32 + mt * 16 + half * 8 + i;
                out[(size_t)mg * N + ng] = acc[mt][nt][i] * s + bb;
            }
        }
    }
}

extern "C" void kernel_launch(void* const* d_in, const int* in_sizes, int n_in,
                              void* d_out, int out_size, void* d_ws, size_t ws_size,
                              hipStream_t stream) {
    const float* x     = (const float*)d_in[0];
    const int*   wq    = (const int*)d_in[1];
    const float* scale = (const float*)d_in[2];
    const float* bias  = (const float*)d_in[3];
    float*       out   = (float*)d_out;

    const int N = in_sizes[2];          // 16384 (scale has one entry per out ch)
    const int K = in_sizes[1] / N;      // 4096
    const int M = in_sizes[0] / K;      // 8192 = B*S

    dim3 grid(N / BN, M / BM);          // 128 x 64 blocks
    dim3 block(256);                    // 8 wave32 waves
    hipLaunchKernelGGL(conv1d_wmma_bf16, grid, block, 0, stream,
                       x, wq, scale, bias, out, M, N, K);
}
